// MessagePassingLayer_49804440764523
// MI455X (gfx1250) — matrix operand order repaired and checked by reading, not compiled, not run on hardware
//
#include <hip/hip_runtime.h>

typedef __attribute__((ext_vector_type(16))) _Float16 v16h;
typedef __attribute__((ext_vector_type(8)))  _Float16 v8h;
typedef __attribute__((ext_vector_type(8)))  float    v8f;

#define NNODES   50000
#define NEDGES   400000
#define NODE_DIM 128
#define EDGE_DIM 64
#define MSG_DIM  128
#define CAT_DIM  320
#define H_STRIDE 328   // halves; 656B/row -> 36-bank rotation, conflict-free A frags

// ---------------- WMMA fragment helpers (contiguous k = 16*half + s order, same for A and B) ----

// 16-bit A 16x32: lane m(=lane&15) holds halves [koff, koff+16),
// koff = kc*32 + (lane>>4)*16, contiguous in a row-major row.
__device__ __forceinline__ v16h load_afrag(const _Float16* arow, int kc, int lhalf) {
  union { v16h v; v8h h[2]; } a;
  const _Float16* p = arow + kc * 32 + lhalf * 16;
  a.h[0] = *(const v8h*)p;
  a.h[1] = *(const v8h*)(p + 8);
  return a.v;
}

// 16-bit B 32x16: lane holds column n = lane&15, halves K = kc*32 + (lane>>4)*16 .. +16,
// contiguous because column n of B == row n of W (we compute h @ W^T).
__device__ __forceinline__ v16h load_bfrag(const _Float16* brow_nt, int kc) {
  union { v16h v; v8h h[2]; } b;
  const _Float16* q = brow_nt + kc * 32;
  b.h[0] = *(const v8h*)q;
  b.h[1] = *(const v8h*)(q + 8);
  return b.v;
}

template <int NT, int KCH>
__device__ __forceinline__ void wmma_gemm(const _Float16* __restrict__ arow,
                                          const _Float16* __restrict__ Wh, int K,
                                          const float* __restrict__ bias,
                                          int lmod, int lhalf, v8f* acc) {
#pragma unroll
  for (int nt = 0; nt < NT; ++nt) {
    float b = bias[nt * 16 + lmod];
    acc[nt] = (v8f){b, b, b, b, b, b, b, b};
  }
#pragma unroll
  for (int kc = 0; kc < KCH; ++kc) {
    v16h a = load_afrag(arow, kc, lhalf);
#pragma unroll
    for (int nt = 0; nt < NT; ++nt) {
      v16h b = load_bfrag(Wh + (size_t)(nt * 16 + lmod) * K + lhalf * 16, kc);
      acc[nt] = __builtin_amdgcn_wmma_f32_16x16x32_f16(false, a, false, b,
                                                       (short)0, acc[nt], false, false);
    }
  }
}

// ---------------- utility kernels ---------------------------------------------------

__global__ void cvt_f32_to_f16_kernel(const float* __restrict__ s,
                                      _Float16* __restrict__ d, int n) {
  int i = blockIdx.x * 256 + threadIdx.x;
  if (i < n) d[i] = (_Float16)s[i];
}

__global__ void zero_f32_kernel(float* __restrict__ p, int n4) {
  int i = blockIdx.x * 256 + threadIdx.x;
  if (i < n4) ((float4*)p)[i] = make_float4(0.f, 0.f, 0.f, 0.f);
}

// ---------------- edge kernel -------------------------------------------------------
// Block = 128 threads (4 waves), 64 edges. LDS row r (328 halves):
//   phase 0: h[r][0..320)            (f16)
//   after fused pass, per-row reuse: m1 in [0,136), m2 in [136,264), e1 in [264,328)
__global__ __launch_bounds__(128) void edge_kernel(
    const float* __restrict__ nodef, const float* __restrict__ edgef,
    const int* __restrict__ eidx,
    const _Float16* __restrict__ Wm1h, const float* __restrict__ bm1,
    const _Float16* __restrict__ Wm2h, const float* __restrict__ bm2,
    const _Float16* __restrict__ Wm3h, const float* __restrict__ bm3,
    const _Float16* __restrict__ We1h, const float* __restrict__ be1,
    const _Float16* __restrict__ We2h, const float* __restrict__ be2,
    float* __restrict__ messages, float* __restrict__ out_edges) {
  __shared__ _Float16 sh[64 * H_STRIDE];
  __shared__ int sidx[2][64];

  const int tid = threadIdx.x;
  const int e0 = blockIdx.x * 64;

  if (tid < 64) {
    sidx[0][tid] = eidx[e0 + tid];           // src
    sidx[1][tid] = eidx[NEDGES + e0 + tid];  // dst
  }
  __syncthreads();

  // Build h = [node[src] | node[dst] | edge] as f16 (float4 gathers; node table is L2-hot).
  for (int i = tid; i < 64 * 80; i += 128) {
    int r  = i / 80;
    int c4 = (i - r * 80) * 4;
    const float* sp;
    if (c4 < 128)      sp = nodef + (size_t)sidx[0][r] * NODE_DIM + c4;
    else if (c4 < 256) sp = nodef + (size_t)sidx[1][r] * NODE_DIM + (c4 - 128);
    else               sp = edgef + (size_t)(e0 + r) * EDGE_DIM + (c4 - 256);
    float4 v = *(const float4*)sp;
    _Float16* dp = &sh[r * H_STRIDE + c4];
    dp[0] = (_Float16)v.x; dp[1] = (_Float16)v.y;
    dp[2] = (_Float16)v.z; dp[3] = (_Float16)v.w;
  }
  __syncthreads();

  const int lane  = tid & 31;
  const int wave  = tid >> 5;
  const int lmod  = lane & 15;
  const int lhalf = lane >> 4;
  _Float16* myrow = &sh[(wave * 16 + lmod) * H_STRIDE];  // A-frag row base for this lane

  // ---- fused pass over K=320: GEMM1 (8 N-tiles, Wm1) + E1 (4 N-tiles, We1) ----
  v8f am[8], ae[4];
#pragma unroll
  for (int nt = 0; nt < 8; ++nt) { float b = bm1[nt * 16 + lmod]; am[nt] = (v8f){b,b,b,b,b,b,b,b}; }
#pragma unroll
  for (int nt = 0; nt < 4; ++nt) { float b = be1[nt * 16 + lmod]; ae[nt] = (v8f){b,b,b,b,b,b,b,b}; }

  const _Float16* bm_base = Wm1h + (size_t)lmod * CAT_DIM + lhalf * 16;
  const _Float16* be_base = We1h + (size_t)lmod * CAT_DIM + lhalf * 16;
#pragma unroll
  for (int kc = 0; kc < 10; ++kc) {
    v16h a = load_afrag(myrow, kc, lhalf);
#pragma unroll
    for (int nt = 0; nt < 8; ++nt) {
      v16h b = load_bfrag(bm_base + (size_t)nt * 16 * CAT_DIM, kc);
      am[nt] = __builtin_amdgcn_wmma_f32_16x16x32_f16(false, a, false, b, (short)0, am[nt], false, false);
    }
#pragma unroll
    for (int nt = 0; nt < 4; ++nt) {
      v16h b = load_bfrag(be_base + (size_t)nt * 16 * CAT_DIM, kc);
      ae[nt] = __builtin_amdgcn_wmma_f32_16x16x32_f16(false, a, false, b, (short)0, ae[nt], false, false);
    }
  }

  // Epilogue: h rows of this wave are dead now; reuse them. C/D layout: lane holds
  // col n = nt*16+lmod; VGPR r holds row m = r + 8*lhalf of the wave's 16-row tile.
#pragma unroll
  for (int nt = 0; nt < 8; ++nt)
#pragma unroll
    for (int r = 0; r < 8; ++r) {
      int m = r + 8 * lhalf;
      float v = am[nt][r]; v = v > 0.f ? v : 0.f;                       // ReLU
      sh[(wave * 16 + m) * H_STRIDE + nt * 16 + lmod] = (_Float16)v;    // m1 -> [0,136)
    }
#pragma unroll
  for (int nt = 0; nt < 4; ++nt)
#pragma unroll
    for (int r = 0; r < 8; ++r) {
      int m = r + 8 * lhalf;
      float v = ae[nt][r]; v = v > 0.f ? v : 0.f;                       // ReLU
      sh[(wave * 16 + m) * H_STRIDE + 264 + nt * 16 + lmod] = (_Float16)v;  // e1
    }

  // ---- GEMM2: m2 = relu(m1 @ Wm2^T + bm2), K=128 ----
  wmma_gemm<8, 4>(myrow, Wm2h, MSG_DIM, bm2, lmod, lhalf, am);
#pragma unroll
  for (int nt = 0; nt < 8; ++nt)
#pragma unroll
    for (int r = 0; r < 8; ++r) {
      int m = r + 8 * lhalf;
      float v = am[nt][r]; v = v > 0.f ? v : 0.f;
      sh[(wave * 16 + m) * H_STRIDE + 136 + nt * 16 + lmod] = (_Float16)v;  // m2
    }

  // ---- GEMM3: m3 = m2 @ Wm3^T + bm3 (no ReLU), scatter-add to messages ----
  wmma_gemm<8, 4>(myrow + 136, Wm3h, MSG_DIM, bm3, lmod, lhalf, am);
#pragma unroll
  for (int nt = 0; nt < 8; ++nt)
#pragma unroll
    for (int r = 0; r < 8; ++r) {
      int m = r + 8 * lhalf;
      int d = sidx[1][wave * 16 + m];
      __hip_atomic_fetch_add(&messages[(size_t)d * MSG_DIM + nt * 16 + lmod],
                             am[nt][r], __ATOMIC_RELAXED, __HIP_MEMORY_SCOPE_AGENT);
    }

  // ---- E2: updated_edges = e1 @ We2^T + be2, K=64 ----
  wmma_gemm<4, 2>(myrow + 264, We2h, EDGE_DIM, be2, lmod, lhalf, ae);
#pragma unroll
  for (int nt = 0; nt < 4; ++nt)
#pragma unroll
    for (int r = 0; r < 8; ++r) {
      int m = r + 8 * lhalf;
      out_edges[(size_t)(e0 + wave * 16 + m) * EDGE_DIM + nt * 16 + lmod] = ae[nt][r];
    }
}

// ---------------- node kernel -------------------------------------------------------
#define A_STRIDE 264  // halves
#define U_STRIDE 129  // floats; 129%64==1 -> conflict-free row reduction
__global__ __launch_bounds__(128) void node_kernel(
    const float* __restrict__ nodef, const float* __restrict__ messages,
    const _Float16* __restrict__ Wnh, const float* __restrict__ bn,
    const float* __restrict__ gamma, const float* __restrict__ beta,
    float* __restrict__ out_nodes) {
  __shared__ _Float16 sa[64 * A_STRIDE];  // 33792 B; aliased as f32 u[64][129] later

  const int tid = threadIdx.x;
  const int n0 = blockIdx.x * 64;

  for (int i = tid; i < 64 * 64; i += 128) {
    int r  = i >> 6;
    int c4 = (i & 63) * 4;
    int n  = n0 + r;
    float4 v = make_float4(0.f, 0.f, 0.f, 0.f);
    if (n < NNODES) {
      v = (c4 < 128) ? *(const float4*)(nodef + (size_t)n * NODE_DIM + c4)
                     : *(const float4*)(messages + (size_t)n * MSG_DIM + (c4 - 128));
    }
    _Float16* dp = &sa[r * A_STRIDE + c4];
    dp[0] = (_Float16)v.x; dp[1] = (_Float16)v.y;
    dp[2] = (_Float16)v.z; dp[3] = (_Float16)v.w;
  }
  __syncthreads();

  const int lane  = tid & 31;
  const int wave  = tid >> 5;
  const int lmod  = lane & 15;
  const int lhalf = lane >> 4;

  v8f acc[8];
  wmma_gemm<8, 8>(&sa[(wave * 16 + lmod) * A_STRIDE], Wnh, NODE_DIM + MSG_DIM, bn,
                  lmod, lhalf, acc);

  __syncthreads();  // all waves done reading sa before aliasing it as f32 u
  float* su = (float*)sa;
#pragma unroll
  for (int nt = 0; nt < 8; ++nt)
#pragma unroll
    for (int r = 0; r < 8; ++r) {
      int m = r + 8 * lhalf;
      float v = acc[nt][r]; v = v > 0.f ? v : 0.f;  // ReLU
      su[(wave * 16 + m) * U_STRIDE + nt * 16 + lmod] = v;
    }
  __syncthreads();

  if (tid < 64) {
    int n = n0 + tid;
    if (n < NNODES) {
      const float* ur = &su[tid * U_STRIDE];
      float s = 0.f, s2 = 0.f;
      for (int c = 0; c < 128; ++c) { float x = ur[c]; s += x; s2 += x * x; }
      float mu  = s * (1.f / 128.f);
      float var = s2 * (1.f / 128.f) - mu * mu;
      var = var > 0.f ? var : 0.f;
      float inv = rsqrtf(var + 1e-5f);
      float* o = out_nodes + (size_t)n * NODE_DIM;
      for (int c = 0; c < 128; ++c)
        o[c] = (ur[c] - mu) * inv * gamma[c] + beta[c];
    }
  }
}

// ---------------- launch ------------------------------------------------------------

extern "C" void kernel_launch(void* const* d_in, const int* in_sizes, int n_in,
                              void* d_out, int out_size, void* d_ws, size_t ws_size,
                              hipStream_t stream) {
  const float* nodef = (const float*)d_in[0];
  const float* edgef = (const float*)d_in[1];
  const int*   eidx  = (const int*)d_in[2];
  const float* Wm1 = (const float*)d_in[3];  const float* bm1 = (const float*)d_in[4];
  const float* Wm2 = (const float*)d_in[5];  const float* bm2 = (const float*)d_in[6];
  const float* Wm3 = (const float*)d_in[7];  const float* bm3 = (const float*)d_in[8];
  const float* We1 = (const float*)d_in[9];  const float* be1 = (const float*)d_in[10];
  const float* We2 = (const float*)d_in[11]; const float* be2 = (const float*)d_in[12];
  const float* Wn  = (const float*)d_in[13]; const float* bn  = (const float*)d_in[14];
  const float* gamma = (const float*)d_in[15];
  const float* beta  = (const float*)d_in[16];

  char* ws = (char*)d_ws;
  _Float16* Wm1h = (_Float16*)(ws + 0);        // 128*320
  _Float16* Wm2h = (_Float16*)(ws + 81920);    // 128*128
  _Float16* Wm3h = (_Float16*)(ws + 114688);   // 128*128
  _Float16* We1h = (_Float16*)(ws + 147456);   // 64*320
  _Float16* We2h = (_Float16*)(ws + 188416);   // 64*64
  _Float16* Wnh  = (_Float16*)(ws + 196608);   // 128*256
  float* messages = (float*)(ws + 262144);     // 50000*128 f32

  auto cvt = [&](const float* s, _Float16* d, int n) {
    cvt_f32_to_f16_kernel<<<(n + 255) / 256, 256, 0, stream>>>(s, d, n);
  };
  cvt(Wm1, Wm1h, 128 * 320);
  cvt(Wm2, Wm2h, 128 * 128);
  cvt(Wm3, Wm3h, 128 * 128);
  cvt(We1, We1h, 64 * 320);
  cvt(We2, We2h, 64 * 64);
  cvt(Wn,  Wnh,  128 * 256);

  const int msg4 = (NNODES * MSG_DIM) / 4;  // 1.6M float4
  zero_f32_kernel<<<(msg4 + 255) / 256, 256, 0, stream>>>(messages, msg4);

  float* out_nodes = (float*)d_out;
  float* out_edges = out_nodes + (size_t)NNODES * NODE_DIM;

  edge_kernel<<<NEDGES / 64, 128, 0, stream>>>(
      nodef, edgef, eidx, Wm1h, bm1, Wm2h, bm2, Wm3h, bm3,
      We1h, be1, We2h, be2, messages, out_edges);

  node_kernel<<<(NNODES + 63) / 64, 128, 0, stream>>>(
      nodef, messages, Wnh, bn, gamma, beta, out_nodes);
}
